// MSGConvNet_74259984548049
// MI455X (gfx1250) — matrix operand order repaired and checked by reading, not compiled, not run on hardware
//
#include <hip/hip_runtime.h>
#include <hip/hip_bf16.h>

typedef __attribute__((ext_vector_type(16))) __bf16        v16bf;
typedef __attribute__((ext_vector_type(8)))  float         v8f;
typedef __attribute__((ext_vector_type(8)))  unsigned int  u32x8;
typedef unsigned short ushort_t;

#define NN    16384
#define EE    262144
#define DD    128
#define KHOP  128
#define NB    16

// ---------------------------------------------------------------- utilities
__global__ void k_fill_u32(unsigned int* p, unsigned int v, int n) {
    int i = blockIdx.x * blockDim.x + threadIdx.x;
    if (i < n) p[i] = v;
}

__device__ __forceinline__ unsigned short f2bf(float f) {
    unsigned int u = __float_as_uint(f);
    unsigned int r = u + 0x7fffu + ((u >> 16) & 1u);   // round-to-nearest-even
    return (unsigned short)(r >> 16);
}

__global__ void k_cvt_bf16(const float* __restrict__ src, ushort_t* __restrict__ dst, int n) {
    int i = blockIdx.x * blockDim.x + threadIdx.x;
    if (i < n) dst[i] = f2bf(src[i]);
}

// dst[n*rows + k] = bf16(src[k*cols + n])   (row-major [K,N] -> bf16 [N,K])
__global__ void k_cvt_bf16_t(const float* __restrict__ src, ushort_t* __restrict__ dst,
                             int rows, int cols) {
    int i = blockIdx.x * blockDim.x + threadIdx.x;
    if (i >= rows * cols) return;
    int k = i % rows, n = i / rows;
    dst[(size_t)n * rows + k] = f2bf(src[(size_t)k * cols + n]);
}

// ---------------------------------------------------------------- graph prep
__global__ void k_init_graph(float* deg, int* cnt, int n) {
    int i = blockIdx.x * blockDim.x + threadIdx.x;
    if (i < n) { deg[i] = 2.0f; cnt[i] = 0; }      // self-loop weight 2
}

__global__ void k_deg_cnt(const int* __restrict__ ei, const float* __restrict__ ew,
                          float* deg, int* cnt, int e) {
    int i = blockIdx.x * blockDim.x + threadIdx.x;
    if (i >= e) return;
    int d = ei[e + i];
    atomicAdd(&deg[d], ew[i]);
    atomicAdd(&cnt[d], 1);
}

__global__ void k_dinv(const float* __restrict__ deg, float* dinv, float* selfn, int n) {
    int i = blockIdx.x * blockDim.x + threadIdx.x;
    if (i >= n) return;
    float g = deg[i];
    float di = (g > 0.0f) ? rsqrtf(g) : 0.0f;
    dinv[i]  = di;
    selfn[i] = di * 2.0f * di;
}

// single-block exclusive scan of cnt[0..n) -> rs[0..n], cursor = rs
__global__ void k_scan(const int* __restrict__ cnt, int* __restrict__ rs,
                       int* __restrict__ cur, int n) {
    __shared__ int part[256];
    __shared__ int excl[257];
    int tid = threadIdx.x;
    int chunk = n / 256;
    int base = tid * chunk;
    int s = 0;
    for (int i = 0; i < chunk; ++i) s += cnt[base + i];
    part[tid] = s;
    __syncthreads();
    if (tid == 0) { int a = 0; for (int i = 0; i < 256; ++i) { excl[i] = a; a += part[i]; } excl[256] = a; }
    __syncthreads();
    int run = excl[tid];
    for (int i = 0; i < chunk; ++i) { rs[base + i] = run; cur[base + i] = run; run += cnt[base + i]; }
    if (tid == 255) rs[n] = excl[256];
}

__global__ void k_csr_fill(const int* __restrict__ ei, const float* __restrict__ ew,
                           const float* __restrict__ dinv, int* __restrict__ cur,
                           int* __restrict__ csrc, float* __restrict__ cnorm, int e) {
    int i = blockIdx.x * blockDim.x + threadIdx.x;
    if (i >= e) return;
    int s = ei[i], d = ei[e + i];
    int idx = atomicAdd(&cur[d], 1);
    csrc[idx]  = s;
    cnorm[idx] = dinv[s] * ew[i] * dinv[d];
}

__global__ void k_batch_count(const int* __restrict__ batch, int* gcount, int n) {
    int i = blockIdx.x * blockDim.x + threadIdx.x;
    if (i < n) atomicAdd(&gcount[batch[i]], 1);
}

__global__ void k_gmeta(const int* __restrict__ gcount, int* gstart, float* invc) {
    if (threadIdx.x != 0 || blockIdx.x != 0) return;
    int a = 0;
    for (int g = 0; g < NB; ++g) {
        gstart[g] = a; a += gcount[g];
        int c = gcount[g] > 0 ? gcount[g] : 1;
        invc[g] = 1.0f / (float)c;
    }
    gstart[NB] = a;
}

// ---------------------------------------------------------------- WMMA GEMM
// C[M,N] = A[M,K](bf16) * Bt[N,K](bf16)^T + bias[N], optional relu. One wave per 16x16 tile.
__global__ void k_wmma_gemm_bf16(const ushort_t* __restrict__ A, const ushort_t* __restrict__ Bt,
                                 const float* __restrict__ bias, float* __restrict__ C,
                                 int M, int N, int K, int relu, int nwaves) {
    int wave = (blockIdx.x * blockDim.x + threadIdx.x) >> 5;
    if (wave >= nwaves) return;                      // wave-uniform: EXEC stays all-ones
    int lane = threadIdx.x & 31;
    int ntiles = N >> 4;
    int mt = wave / ntiles, nt = wave % ntiles;
    int half = lane >> 4;
    int l16  = lane & 15;
    int m = mt * 16 + l16;
    int n = nt * 16 + l16;

    v8f acc;
    float bv = bias[n];
#pragma unroll
    for (int j = 0; j < 8; ++j) acc[j] = bv;

    const ushort_t* arow = A  + (size_t)m * K;
    const ushort_t* brow = Bt + (size_t)n * K;
    for (int k0 = 0; k0 < K; k0 += 32) {
        int ka = k0 + half * 8;    // A: lanes 0-15 K=k0..+7/+16..+23, lanes 16-31 shifted by 8
        int kb = k0 + half * 8;    // B: lanes 0-15 K=k0..+7/+16..+23, lanes 16-31 shifted by 8
        uint4 a0 = *(const uint4*)(arow + ka);
        uint4 a1 = *(const uint4*)(arow + ka + 16);
        uint4 b0 = *(const uint4*)(brow + kb);
        uint4 b1 = *(const uint4*)(brow + kb + 16);
        u32x8 au, bu;
        au[0]=a0.x; au[1]=a0.y; au[2]=a0.z; au[3]=a0.w;
        au[4]=a1.x; au[5]=a1.y; au[6]=a1.z; au[7]=a1.w;
        bu[0]=b0.x; bu[1]=b0.y; bu[2]=b0.z; bu[3]=b0.w;
        bu[4]=b1.x; bu[5]=b1.y; bu[6]=b1.z; bu[7]=b1.w;
        v16bf af = __builtin_bit_cast(v16bf, au);
        v16bf bf = __builtin_bit_cast(v16bf, bu);
        acc = __builtin_amdgcn_wmma_f32_16x16x32_bf16(false, af, false, bf,
                                                      (short)0, acc, false, false);
    }
    if (relu) {
#pragma unroll
        for (int j = 0; j < 8; ++j) acc[j] = fmaxf(acc[j], 0.0f);
    }
#pragma unroll
    for (int j = 0; j < 8; ++j) {
        int mr = half ? (j + 8) : j;                  // C layout: VGPR j -> M=j (+8 upper lanes)
        C[(size_t)(mt * 16 + mr) * N + nt * 16 + l16] = acc[j];
    }
}

// ---------------------------------------------------------------- propagation hop
__global__ void k_hop(const float* __restrict__ hin, float* __restrict__ hout,
                      const int* __restrict__ rs, const int* __restrict__ csrc,
                      const float* __restrict__ cnorm, const float* __restrict__ selfn, int n) {
    int node = (blockIdx.x * blockDim.x + threadIdx.x) >> 5;
    if (node >= n) return;
    int lane = threadIdx.x & 31;
    int d = lane << 2;                                 // 4 features per lane
    const float4 hs = *(const float4*)(hin + (size_t)node * DD + d);
    float sn = selfn[node];
    float ax = sn * hs.x, ay = sn * hs.y, az = sn * hs.z, aw = sn * hs.w;
    int beg = rs[node], end = rs[node + 1];
    int j = beg;
    int   s = (j < end) ? csrc[j]  : 0;
    float w = (j < end) ? cnorm[j] : 0.0f;
    while (j < end) {
        int jn = j + 1;
        int   s2 = (jn < end) ? csrc[jn]  : 0;
        float w2 = (jn < end) ? cnorm[jn] : 0.0f;
        if (jn < end) __builtin_prefetch(hin + (size_t)s2 * DD + d, 0, 3); // global_prefetch_b8
        const float4 v = *(const float4*)(hin + (size_t)s * DD + d);
        ax += w * v.x; ay += w * v.y; az += w * v.z; aw += w * v.w;
        s = s2; w = w2; j = jn;
    }
    float4 o; o.x = ax; o.y = ay; o.z = az; o.w = aw;
    *(float4*)(hout + (size_t)node * DD + d) = o;
}

// pooled[g*128+d] += invc[g]*gamma[d]*sum_{n in graph g, chunk c} h[n,d]; grid (16,8), block 128
__global__ void k_pool(const float* __restrict__ h, const int* __restrict__ gstart,
                       const float* __restrict__ gam, const float* __restrict__ invc,
                       float* __restrict__ pooled) {
    int g = blockIdx.x, c = blockIdx.y, d = threadIdx.x;
    int beg = gstart[g] + c, end = gstart[g + 1];
    float s = 0.0f;
    for (int n0 = beg; n0 < end; n0 += 8) s += h[(size_t)n0 * DD + d];
    atomicAdd(&pooled[g * DD + d], s * gam[d] * invc[g]);
}

// ---------------------------------------------------------------- image path
__global__ void k_img(const float* __restrict__ pooled, float* __restrict__ img) {
    int idx = blockIdx.x * blockDim.x + threadIdx.x;
    if (idx >= NB * KHOP * DD) return;
    int d = idx & 127, t = idx >> 7;
    int k = t & 127, b = t >> 7;
    img[idx] = pooled[(size_t)k * (NB * DD) + b * DD + d];  // img[b][k][d] = pooled[k][b][d]
}

// Conv3x3(SAME)+bias -> ReLU -> BN(scale,beta) -> MaxPool2x2 ; one thread per pooled output
__global__ void k_conv_block(const float* __restrict__ in, const float* __restrict__ wt,
                             const float* __restrict__ bias, const float* __restrict__ bng,
                             const float* __restrict__ bnb, float* __restrict__ out,
                             int Bn, int Cin, int Cout, int H, int W) {
    int H2 = H >> 1, W2 = W >> 1;
    int total = Bn * Cout * H2 * W2;
    int idx = blockIdx.x * blockDim.x + threadIdx.x;
    if (idx >= total) return;
    int x2 = idx % W2; int t = idx / W2;
    int y2 = t % H2;   t /= H2;
    int co = t % Cout; int b = t / Cout;
    float scale = bng[co] * rsqrtf(1.0f + 1e-5f);
    float beta  = bnb[co];
    float bsv   = bias[co];
    float mx = -3.402823466e38f;
    for (int dy = 0; dy < 2; ++dy)
    for (int dx = 0; dx < 2; ++dx) {
        int y = y2 * 2 + dy, x = x2 * 2 + dx;
        float s = bsv;
        for (int ci = 0; ci < Cin; ++ci) {
            const float* ip = in + ((size_t)b * Cin + ci) * H * W;
            const float* wp = wt + ((size_t)co * Cin + ci) * 9;
            for (int ky = 0; ky < 3; ++ky) {
                int yy = y + ky - 1;
                if (yy < 0 || yy >= H) continue;
                for (int kx = 0; kx < 3; ++kx) {
                    int xx = x + kx - 1;
                    if (xx < 0 || xx >= W) continue;
                    s += ip[yy * W + xx] * wp[ky * 3 + kx];
                }
            }
        }
        float v = fmaxf(s, 0.0f) * scale + beta;
        mx = fmaxf(mx, v);
    }
    out[idx] = mx;
}

// ---------------------------------------------------------------- fc3 + softmax (wave per row)
__global__ void k_fc3_softmax(const float* __restrict__ z, const float* __restrict__ w,
                              const float* __restrict__ b, float* __restrict__ out) {
    int row = blockIdx.x;
    int lane = threadIdx.x;
    float logit[10];
#pragma unroll
    for (int c = 0; c < 10; ++c) {
        float p = 0.0f;
        for (int k = lane; k < 2048; k += 32) p += z[(size_t)row * 2048 + k] * w[(size_t)k * 10 + c];
#pragma unroll
        for (int off = 16; off; off >>= 1) p += __shfl_xor(p, off, 32);
        logit[c] = p + b[c];
    }
    float m = logit[0];
#pragma unroll
    for (int c = 1; c < 10; ++c) m = fmaxf(m, logit[c]);
    float e[10], sum = 0.0f;
#pragma unroll
    for (int c = 0; c < 10; ++c) { e[c] = __expf(logit[c] - m); sum += e[c]; }
    if (lane == 0) {
#pragma unroll
        for (int c = 0; c < 10; ++c) out[row * 10 + c] = e[c] / sum;
    }
}

// ================================================================ launcher
extern "C" void kernel_launch(void* const* d_in, const int* in_sizes, int n_in,
                              void* d_out, int out_size, void* d_ws, size_t ws_size,
                              hipStream_t stream) {
    const float* x      = (const float*)d_in[0];
    const int*   ei     = (const int*)  d_in[1];
    const float* ew     = (const float*)d_in[2];
    const int*   batch  = (const int*)  d_in[3];
    const float* W      = (const float*)d_in[4];
    const float* b_lin  = (const float*)d_in[5];
    const float* gamma  = (const float*)d_in[6];
    const float* c1w = (const float*)d_in[7];  const float* c1b = (const float*)d_in[8];
    const float* g1  = (const float*)d_in[9];  const float* be1 = (const float*)d_in[10];
    const float* c2w = (const float*)d_in[11]; const float* c2b = (const float*)d_in[12];
    const float* g2  = (const float*)d_in[13]; const float* be2 = (const float*)d_in[14];
    const float* c3w = (const float*)d_in[15]; const float* c3b = (const float*)d_in[16];
    const float* g3  = (const float*)d_in[17]; const float* be3 = (const float*)d_in[18];
    const float* c4w = (const float*)d_in[19]; const float* c4b = (const float*)d_in[20];
    const float* g4  = (const float*)d_in[21]; const float* be4 = (const float*)d_in[22];
    const float* f1w = (const float*)d_in[23]; const float* f1b = (const float*)d_in[24];
    const float* f2w = (const float*)d_in[25]; const float* f2b = (const float*)d_in[26];
    const float* f3w = (const float*)d_in[27]; const float* f3b = (const float*)d_in[28];
    float* out = (float*)d_out;

    // ---------------- workspace layout (persistent + graph/FC union)
    char* ws = (char*)d_ws;
    size_t off = 0;
    auto alloc = [&](size_t bytes) -> char* {
        char* p = ws + off; off += (bytes + 255) & ~(size_t)255; return p;
    };
    float* pooled = (float*)alloc((size_t)KHOP * NB * DD * 4);
    int*   gcount = (int*)  alloc(NB * 4);
    int*   gstart = (int*)  alloc((NB + 1) * 4);
    float* invc   = (float*)alloc(NB * 4);
    float* img = (float*)alloc((size_t)NB * 1 * 128 * 128 * 4);
    float* z1  = (float*)alloc((size_t)NB * 8  * 64 * 64 * 4);
    float* z2  = (float*)alloc((size_t)NB * 16 * 32 * 32 * 4);
    float* z3  = (float*)alloc((size_t)NB * 32 * 16 * 16 * 4);
    float* z4  = (float*)alloc((size_t)NB * 64 * 8  * 8  * 4);
    float*    zfc1  = (float*)   alloc(16 * 2048 * 4);
    float*    zfc2  = (float*)   alloc(16 * 2048 * 4);
    ushort_t* actb  = (ushort_t*)alloc(16 * 4096 * 2);
    ushort_t* act2b = (ushort_t*)alloc(16 * 2048 * 2);
    size_t ubase = off;
    // graph-phase view of union region
    float* deg   = (float*)alloc(NN * 4);
    float* dinv  = (float*)alloc(NN * 4);
    float* selfn = (float*)alloc(NN * 4);
    int*   cnt   = (int*)  alloc(NN * 4);
    int*   rs    = (int*)  alloc((NN + 1) * 4);
    int*   cur   = (int*)  alloc(NN * 4);
    int*   csrc  = (int*)  alloc((size_t)EE * 4);
    float* cnorm = (float*)alloc((size_t)EE * 4);
    ushort_t* xb  = (ushort_t*)alloc((size_t)NN * DD * 2);
    ushort_t* wtb = (ushort_t*)alloc((size_t)DD * DD * 2);
    float* ha = (float*)alloc((size_t)NN * DD * 4);
    float* hb = (float*)alloc((size_t)NN * DD * 4);
    // FC-phase view (graph buffers dead by then)
    ushort_t* wt1b = (ushort_t*)(ws + ubase);                 // [2048,4096] bf16 = 16 MB
    ushort_t* wt2b = wt1b + (size_t)2048 * 4096;              // [2048,2048] bf16 = 8 MB

    const int T = 256;
    auto blocks = [](long n, int t) { return (int)((n + t - 1) / t); };

    // ---------------- graph preprocessing (rebuilt every call: deterministic)
    k_fill_u32<<<blocks(KHOP * NB * DD, T), T, 0, stream>>>((unsigned*)pooled, 0u, KHOP * NB * DD);
    k_fill_u32<<<1, 32, 0, stream>>>((unsigned*)gcount, 0u, NB);
    k_init_graph<<<blocks(NN, T), T, 0, stream>>>(deg, cnt, NN);
    k_deg_cnt<<<blocks(EE, T), T, 0, stream>>>(ei, ew, deg, cnt, EE);
    k_dinv<<<blocks(NN, T), T, 0, stream>>>(deg, dinv, selfn, NN);
    k_scan<<<1, 256, 0, stream>>>(cnt, rs, cur, NN);
    k_csr_fill<<<blocks(EE, T), T, 0, stream>>>(ei, ew, dinv, cur, csrc, cnorm, EE);
    k_batch_count<<<blocks(NN, T), T, 0, stream>>>(batch, gcount, NN);
    k_gmeta<<<1, 1, 0, stream>>>(gcount, gstart, invc);

    // ---------------- h0 = x @ W + b  (bf16 WMMA)
    k_cvt_bf16<<<blocks((long)NN * DD, T), T, 0, stream>>>(x, xb, NN * DD);
    k_cvt_bf16_t<<<blocks((long)DD * DD, T), T, 0, stream>>>(W, wtb, DD, DD);
    {
        int nwaves = (NN / 16) * (DD / 16);
        k_wmma_gemm_bf16<<<blocks((long)nwaves * 32, T), T, 0, stream>>>(
            xb, wtb, b_lin, ha, NN, DD, DD, 0, nwaves);
    }

    // ---------------- K=128 propagation hops + per-hop pooling
    float* hcur = ha; float* hnxt = hb;
    for (int hop = 0; hop < KHOP; ++hop) {
        k_hop<<<blocks((long)NN * 32, T), T, 0, stream>>>(hcur, hnxt, rs, csrc, cnorm, selfn, NN);
        dim3 pg(NB, 8);
        k_pool<<<pg, DD, 0, stream>>>(hnxt, gstart, gamma + hop * DD, invc,
                                      pooled + (size_t)hop * NB * DD);
        float* t = hcur; hcur = hnxt; hnxt = t;
    }

    // ---------------- conv net
    k_img<<<blocks((long)NB * KHOP * DD, T), T, 0, stream>>>(pooled, img);
    k_conv_block<<<blocks((long)NB * 8  * 64 * 64, T), T, 0, stream>>>(img, c1w, c1b, g1, be1, z1, NB, 1,  8,  128, 128);
    k_conv_block<<<blocks((long)NB * 16 * 32 * 32, T), T, 0, stream>>>(z1,  c2w, c2b, g2, be2, z2, NB, 8,  16, 64,  64);
    k_conv_block<<<blocks((long)NB * 32 * 16 * 16, T), T, 0, stream>>>(z2,  c3w, c3b, g3, be3, z3, NB, 16, 32, 32,  32);
    k_conv_block<<<blocks((long)NB * 64 * 8  * 8,  T), T, 0, stream>>>(z3,  c4w, c4b, g4, be4, z4, NB, 32, 64, 16,  16);

    // ---------------- FC head (bf16 WMMA)
    k_cvt_bf16<<<blocks(16L * 4096, T), T, 0, stream>>>(z4, actb, 16 * 4096);
    k_cvt_bf16_t<<<blocks(4096L * 2048, T), T, 0, stream>>>(f1w, wt1b, 4096, 2048);
    {
        int nwaves = 1 * (2048 / 16);
        k_wmma_gemm_bf16<<<blocks((long)nwaves * 32, T), T, 0, stream>>>(
            actb, wt1b, f1b, zfc1, 16, 2048, 4096, 1, nwaves);
    }
    k_cvt_bf16<<<blocks(16L * 2048, T), T, 0, stream>>>(zfc1, act2b, 16 * 2048);
    k_cvt_bf16_t<<<blocks(2048L * 2048, T), T, 0, stream>>>(f2w, wt2b, 2048, 2048);
    {
        int nwaves = 1 * (2048 / 16);
        k_wmma_gemm_bf16<<<blocks((long)nwaves * 32, T), T, 0, stream>>>(
            act2b, wt2b, f2b, zfc2, 16, 2048, 2048, 1, nwaves);
    }
    k_fc3_softmax<<<NB, 32, 0, stream>>>(zfc2, f3w, f3b, out);

    (void)in_sizes; (void)n_in; (void)out_size; (void)ws_size;
}
